// Translator_NN_caps_73169062855102
// MI455X (gfx1250) — matrix operand order repaired and checked by reading, not compiled, not run on hardware
//
#include <hip/hip_runtime.h>

// ---------------------------------------------------------------------------
// Problem constants (match reference module)
// ---------------------------------------------------------------------------
#define B_       32
#define E_       8
#define CIN      1024
#define NUM_CAPS 256
#define COUT     864
#define DEPTH    384
#define LIGHT    48
#define M_TOTAL  (B_ * NUM_CAPS * E_)   // 65536 GEMM rows

// GEMM tiling: workgroup = 8 waves (256 thr, wave32). WG tile 128x96,
// wave tile 32x48 (= 2x3 WMMA 16x16 tiles), K chunk 32 (one f16 WMMA K).
#define MTILE  128
#define NTILE  96
#define KTILE  32
#define LDSK   36   // padded K stride in f32: 144 B (16B aligned; 36 mod 64 banks -> conflict-free rows)

typedef __attribute__((ext_vector_type(16))) _Float16 v16h;
typedef __attribute__((ext_vector_type(8)))  float    v8f;
typedef int v4i_ __attribute__((vector_size(16)));   // pointee type the async builtin expects

static __device__ __forceinline__ v8f wmma_f16(v16h a, v16h b, v8f c) {
  // D = A*B + C, f32 accumulate. 8-arg form: (neg_a, A, neg_b, B, c_mod, C, reuse_a, reuse_b)
  return __builtin_amdgcn_wmma_f32_16x16x32_f16(false, a, false, b, (short)0, c, false, false);
}

// --------------------------- async copy plumbing ---------------------------
#if __has_builtin(__builtin_amdgcn_global_load_async_to_lds_b128)
#define HAVE_ASYNC 1
// GLOBAL_LOAD_ASYNC_TO_LDS_B128: 16B global -> LDS, tracked by ASYNCcnt.
// Signature (from compiler diagnostic): (int4 AS1*, int4 AS3*, imm offset, imm cpol)
#define ASYNC_B128(g, l)                                                     \
  __builtin_amdgcn_global_load_async_to_lds_b128(                            \
      (__attribute__((address_space(1))) v4i_*)(g),                          \
      (__attribute__((address_space(3))) v4i_*)(l), 0, 0)
#if __has_builtin(__builtin_amdgcn_s_wait_asynccnt)
#define WAIT_ASYNC(n) __builtin_amdgcn_s_wait_asynccnt(n)
#else
#define WAIT_ASYNC(n) asm volatile("s_wait_asynccnt %0" ::"i"(n) : "memory")
#endif
#else
#define HAVE_ASYNC 0
// Fallback: synchronous VGPR round-trip copy (compiles everywhere).
#define ASYNC_B128(g, l) (*(float4*)(l) = *(const float4*)(g))
#define WAIT_ASYNC(n) ((void)0)
#endif

// ---------------------------------------------------------------------------
// Kernel 1: split-f16 WMMA GEMM   feat = X @ W^T + b ;  map = feat * basis
//   X: [M_TOTAL, CIN] f32 (row m = ((b*256+cap)*8+e)),  W: [COUT, CIN] f32.
// Raw f32 tiles are DMA'd to LDS asynchronously (double buffered); the hi/lo
// f16 split happens at fragment-build time so C += Ah*Bh + Ah*Bl + Al*Bh
// gives ~f32 accuracy at 3 f16-WMMAs per K chunk (2.7x the f32 WMMA rate).
// ---------------------------------------------------------------------------
__global__ __launch_bounds__(256)
void gemm_feat_map_kernel(const float* __restrict__ X,
                          const float* __restrict__ W,
                          const float* __restrict__ bias,
                          const float* __restrict__ basis,   // [NUM_CAPS*E_, COUT]
                          float* __restrict__ feat_ws,       // [M_TOTAL, COUT]
                          float* __restrict__ attn_out)      // [M_TOTAL, COUT]
{
  __shared__ __align__(16) float Af[2][MTILE][LDSK];   // 36,864 B
  __shared__ __align__(16) float Bf[2][NTILE][LDSK];   // 27,648 B   (total 64,512 B)

  const int t    = threadIdx.x;
  const int lane = t & 31;
  const int wv   = t >> 5;
  const int wm   = wv & 3;            // M sub-block (32 rows each)
  const int wn   = wv >> 2;           // N sub-block (48 cols each)
  const int m_base = blockIdx.y * MTILE;
  const int o_base = blockIdx.x * NTILE;

  v8f acc[2][3];
  #pragma unroll
  for (int i = 0; i < 2; ++i)
    #pragma unroll
    for (int j = 0; j < 3; ++j)
      #pragma unroll
      for (int r = 0; r < 8; ++r) acc[i][j][r] = 0.0f;

  // ---- stage one K chunk: 7 async B128 per thread (A: 4, B: 3)
  auto stage = [&](int buf, int k0) {
    #pragma unroll
    for (int i = 0; i < 4; ++i) {                 // A tile: 128x32 f32 = 1024 f4
      const int idx = i * 256 + t;
      const int row = idx >> 3;
      const int kq  = idx & 7;
      ASYNC_B128(X + (size_t)(m_base + row) * CIN + k0 + kq * 4,
                 &Af[buf][row][kq * 4]);
    }
    #pragma unroll
    for (int i = 0; i < 3; ++i) {                 // B tile: 96x32 f32 = 768 f4
      const int idx = i * 256 + t;
      const int row = idx >> 3;
      const int kq  = idx & 7;
      ASYNC_B128(W + (size_t)(o_base + row) * CIN + k0 + kq * 4,
                 &Bf[buf][row][kq * 4]);
    }
  };

  stage(0, 0);                                    // prologue

  for (int k0 = 0; k0 < CIN; k0 += KTILE) {
    const int  buf  = (k0 / KTILE) & 1;
    const bool more = (k0 + KTILE) < CIN;
    if (more) stage(buf ^ 1, k0 + KTILE);         // overlap next HBM fetch
    if (more) { WAIT_ASYNC(7); } else { WAIT_ASYNC(0); }  // current buf landed
    __syncthreads();                              // ...in every wave

    // ---- A fragments (ISA f16 A 16x32 layout):
    // lanes 0-15 : row = lane,    K[0..7] + K[16..23]
    // lanes 16-31: row = lane-16, K[8..15] + K[24..31]
    const int arow = lane & 15;
    const int asel = (lane >> 4) * 8;             // K sub-offset 0 / 8
    v16h afh[2], afl[2];
    #pragma unroll
    for (int mt = 0; mt < 2; ++mt) {
      const float* rp = &Af[buf][wm * 32 + mt * 16 + arow][0];
      const float4 q0 = *reinterpret_cast<const float4*>(rp + asel);
      const float4 q1 = *reinterpret_cast<const float4*>(rp + asel + 4);
      const float4 q2 = *reinterpret_cast<const float4*>(rp + 16 + asel);
      const float4 q3 = *reinterpret_cast<const float4*>(rp + 16 + asel + 4);
      const float vals[16] = {q0.x, q0.y, q0.z, q0.w, q1.x, q1.y, q1.z, q1.w,
                              q2.x, q2.y, q2.z, q2.w, q3.x, q3.y, q3.z, q3.w};
      #pragma unroll
      for (int i = 0; i < 16; ++i) {
        const _Float16 hi = (_Float16)vals[i];
        afh[mt][i] = hi;
        afl[mt][i] = (_Float16)(vals[i] - (float)hi);   // -> v_fma_mix + cvt
      }
    }
    // ---- B fragments (ISA f16 B 32x16 layout):
    // lanes 0-15: col = lane, K[0..7] + K[16..23] ; lanes 16-31: col = lane-16, K[8..15] + K[24..31]
    const int bcol = lane & 15;
    const int bsel = (lane >> 4) * 8;             // K sub-offset 0 / 8
    v16h bfh[3], bfl[3];
    #pragma unroll
    for (int nt = 0; nt < 3; ++nt) {
      const float* rp = &Bf[buf][wn * 48 + nt * 16 + bcol][0];
      const float4 q0 = *reinterpret_cast<const float4*>(rp + bsel);
      const float4 q1 = *reinterpret_cast<const float4*>(rp + bsel + 4);
      const float4 q2 = *reinterpret_cast<const float4*>(rp + 16 + bsel);
      const float4 q3 = *reinterpret_cast<const float4*>(rp + 16 + bsel + 4);
      const float vals[16] = {q0.x, q0.y, q0.z, q0.w, q1.x, q1.y, q1.z, q1.w,
                              q2.x, q2.y, q2.z, q2.w, q3.x, q3.y, q3.z, q3.w};
      #pragma unroll
      for (int i = 0; i < 16; ++i) {
        const _Float16 hi = (_Float16)vals[i];
        bfh[nt][i] = hi;
        bfl[nt][i] = (_Float16)(vals[i] - (float)hi);
      }
    }

    // ---- 2x3 tiles x 3 split terms = 18 WMMAs per K chunk per wave
    #pragma unroll
    for (int mt = 0; mt < 2; ++mt)
      #pragma unroll
      for (int nt = 0; nt < 3; ++nt) {
        acc[mt][nt] = wmma_f16(afh[mt], bfh[nt], acc[mt][nt]);
        acc[mt][nt] = wmma_f16(afh[mt], bfl[nt], acc[mt][nt]);
        acc[mt][nt] = wmma_f16(afl[mt], bfh[nt], acc[mt][nt]);
      }
    __syncthreads();          // frag reads done before next overwrite of buf^1
  }

  // ---- epilogue: C 16x16 f32 layout: VGPR r -> (M=r, N=lane) lanes 0-15,
  // (M=8+r, N=lane-16) lanes 16-31. Fuse +bias and *basis.
  const int ccol = lane & 15;
  const int rsel = (lane >> 4) * 8;
  #pragma unroll
  for (int mt = 0; mt < 2; ++mt) {
    #pragma unroll
    for (int nt = 0; nt < 3; ++nt) {
      const int o  = o_base + wn * 48 + nt * 16 + ccol;
      const float bo = bias[o];
      #pragma unroll
      for (int r = 0; r < 8; ++r) {
        const int    gm  = m_base + wm * 32 + mt * 16 + rsel + r;
        const float  f   = acc[mt][nt][r] + bo;
        const float  bs  = basis[(size_t)(gm & (NUM_CAPS * E_ - 1)) * COUT + o]; // (cap*8+e)
        const size_t idx = (size_t)gm * COUT + o;
        feat_ws[idx]  = f;
        attn_out[idx] = f * bs;
      }
    }
  }
}

// ---------------------------------------------------------------------------
// Kernel 2: online softmax stats over the caps axis (stride E_*COUT rows)
// one thread per (b, e, o)  -> max, sum(exp), argmax
// ---------------------------------------------------------------------------
__global__ __launch_bounds__(256)
void softmax_stats_kernel(const float* __restrict__ attn,
                          float* __restrict__ maxv,
                          float* __restrict__ sumexp,
                          int*   __restrict__ amax)
{
  const int tid = blockIdx.x * 256 + threadIdx.x;
  if (tid >= B_ * E_ * COUT) return;
  const int o  = tid % COUT;
  const int be = tid / COUT;              // b*8 + e
  const int b  = be >> 3;
  const int e  = be & 7;
  const size_t stride = (size_t)E_ * COUT;                                  // 6912
  const size_t base   = ((size_t)b * NUM_CAPS * E_ + e) * COUT + o;         // cap = 0

  float m = attn[base];
  float s = 1.0f;
  int   am = 0;
  for (int c = 1; c < NUM_CAPS; ++c) {
    const float v = attn[base + (size_t)c * stride];
    if (v > m) { s = s * __expf(m - v) + 1.0f; m = v; am = c; }
    else       { s += __expf(v - m); }
  }
  maxv[tid] = m; sumexp[tid] = s; amax[tid] = am;
}

// ---------------------------------------------------------------------------
// Kernel 3: output[b,cap,o] = sum_e softmax * feat * mask
//   mask: o<384 -> onehot(argmax[o]); 384<=o<768 -> onehot(argmax[o-384]);
//         o>=768 -> 1/NUM_CAPS
// ---------------------------------------------------------------------------
__global__ __launch_bounds__(256)
void finalize_kernel(const float* __restrict__ attn,
                     const float* __restrict__ feat,
                     const float* __restrict__ maxv,
                     const float* __restrict__ sumexp,
                     const int*   __restrict__ amax,
                     float* __restrict__ out)
{
  const int tid = blockIdx.x * 256 + threadIdx.x;
  if (tid >= B_ * NUM_CAPS * COUT) return;
  const int o   = tid % COUT;
  const int bc  = tid / COUT;             // b*256 + cap
  const int cap = bc & (NUM_CAPS - 1);
  const int b   = bc >> 8;
  const int wcol = (o < 2 * DEPTH) ? ((o < DEPTH) ? o : o - DEPTH) : -1;

  float acc = 0.0f;
  #pragma unroll
  for (int e = 0; e < E_; ++e) {
    const size_t idx = ((size_t)bc * E_ + e) * COUT + o;
    const int    s   = (b * E_ + e) * COUT;
    const float  fin = __expf(attn[idx] - maxv[s + o]) / sumexp[s + o];
    const float  wgt = (wcol >= 0) ? ((amax[s + wcol] == cap) ? 1.0f : 0.0f)
                                   : (1.0f / NUM_CAPS);
    acc += fin * feat[idx] * wgt;
  }
  out[tid] = acc;
}

// ---------------------------------------------------------------------------
extern "C" void kernel_launch(void* const* d_in, const int* in_sizes, int n_in,
                              void* d_out, int out_size, void* d_ws, size_t ws_size,
                              hipStream_t stream) {
  (void)in_sizes; (void)n_in; (void)out_size; (void)ws_size;
  const float* X     = (const float*)d_in[0];   // x        [B,16,16,E,CIN]
  const float* W     = (const float*)d_in[1];   // W        [COUT,CIN]
  const float* bias  = (const float*)d_in[2];   // b        [COUT]
  const float* basis = (const float*)d_in[3];   // basis    [1,256,E,COUT]

  float* out      = (float*)d_out;                                   // [B,256,COUT]
  float* attn_out = out + (size_t)B_ * NUM_CAPS * COUT;              // [B,256,E,COUT]

  // workspace: feat (226.5 MB) | maxv | sumexp | amax
  float* feat   = (float*)d_ws;
  const size_t nfeat  = (size_t)M_TOTAL * COUT;
  const size_t nstats = (size_t)B_ * E_ * COUT;
  float* maxv   = feat + nfeat;
  float* sumexp = maxv + nstats;
  int*   amax   = (int*)(sumexp + nstats);

  dim3 g1(COUT / NTILE, M_TOTAL / MTILE);   // (9, 512), 256 threads = 8 waves
  gemm_feat_map_kernel<<<g1, 256, 0, stream>>>(X, W, bias, basis, feat, attn_out);

  softmax_stats_kernel<<<(int)((nstats + 255) / 256), 256, 0, stream>>>(
      attn_out, maxv, sumexp, amax);

  finalize_kernel<<<(B_ * NUM_CAPS * COUT + 255) / 256, 256, 0, stream>>>(
      attn_out, feat, maxv, sumexp, amax, out);
}
